// SegTransformerDecoder_49967649522152
// MI455X (gfx1250) — matrix operand order repaired and checked by reading, not compiled, not run on hardware
//
#include <hip/hip_runtime.h>
#include <hip/hip_bf16.h>
#include <cstdint>

// ---------------------------------------------------------------------------
// MI455X (gfx1250) implementation of SegTransformerDecoder block.
// Convs -> implicit GEMM on v_wmma_f32_16x16x32_f16 (f16 in, f32 accumulate),
// 128(M) x 64(N) block tile, 4 WMMAs per K-step per wave.
// 1x1 convs: B tiles double-buffer-staged by the Tensor Data Mover
// (tensor_load_to_lds, TENSORcnt pipelining: issue next tile, wait <=1).
// Spatial convs: gather staging (im2col duplicates data) + global_prefetch_b8
// on the weight stream. Sampling stage is a VALU/gather kernel.
// ---------------------------------------------------------------------------

typedef __attribute__((ext_vector_type(16))) _Float16 v16h;
typedef __attribute__((ext_vector_type(8)))  _Float16 v8h;
typedef __attribute__((ext_vector_type(8)))  float    v8f;
typedef __attribute__((ext_vector_type(4)))  unsigned u32x4;
typedef __attribute__((ext_vector_type(4)))  int      i32x4;
typedef __attribute__((ext_vector_type(8)))  int      i32x8;

#define HW_N 10000   // 100x100 spatial positions

#if defined(__has_builtin)
#  if __has_builtin(__builtin_amdgcn_tensor_load_to_lds)
#    define HAVE_TDM 1
#  else
#    define HAVE_TDM 0
#  endif
#else
#  define HAVE_TDM 0
#endif

__device__ __forceinline__ float gelu_exact(float x) {
    return 0.5f * x * (1.0f + erff(x * 0.70710678118654752f));
}

#if HAVE_TDM
// 2D TDM load: tile_h x tile_w f16 elements from a row-major tensor
// (row stride = stride_elems) into LDS at lds_byte_off (k-major, packed rows).
// D# per CDNA5 ISA 8.3/8.4.
__device__ __forceinline__ void tdm_load_tile_f16(
    unsigned lds_byte_off, const _Float16* gptr,
    unsigned tensor_w, unsigned tensor_h, unsigned stride_elems,
    unsigned tile_w, unsigned tile_h)
{
    unsigned long long ga = (unsigned long long)(uintptr_t)gptr;
    u32x4 g0;
    g0[0] = 1u;                                            // count=1, user mode
    g0[1] = lds_byte_off;                                  // lds_addr (bytes)
    g0[2] = (unsigned)ga;                                  // global_addr lo
    g0[3] = (unsigned)((ga >> 32) & 0x01ffffffu) | (2u << 30); // addr hi | type=2
    i32x8 g1;
    g1[0] = (int)(1u << 16);                               // data_size = 1 (2 B)
    g1[1] = (int)((tensor_w & 0xffffu) << 16);             // tensor_dim0[15:0]
    g1[2] = (int)((tensor_w >> 16) | ((tensor_h & 0xffffu) << 16)); // d0 hi | d1 lo
    g1[3] = (int)((tensor_h >> 16) | (tile_w << 16));      // d1 hi | tile_dim0
    g1[4] = (int)tile_h;                                   // tile_dim1 (tile_dim2=0)
    g1[5] = (int)stride_elems;                             // tensor_dim0_stride lo
    g1[6] = 0;
    g1[7] = 0;
    i32x4 z4 = {0, 0, 0, 0};
#if __clang_major__ >= 23
    i32x8 z8 = {0, 0, 0, 0, 0, 0, 0, 0};
    __builtin_amdgcn_tensor_load_to_lds(g0, g1, z4, z4, z8, 0);
#else
    __builtin_amdgcn_tensor_load_to_lds(g0, g1, z4, z4, 0);
#endif
}
#endif // HAVE_TDM

// -------------------------------- k_cvt ------------------------------------
__global__ __launch_bounds__(256) void k_cvt(const float* __restrict__ src,
                                             _Float16* __restrict__ dst,
                                             int n, int ntot) {
    int i = blockIdx.x * 256 + threadIdx.x;
    if (i < ntot) dst[i] = (i < n) ? (_Float16)src[i] : (_Float16)0.0f;
}

// -------------------------------- k_pad ------------------------------------
__global__ __launch_bounds__(256) void k_pad(const float* __restrict__ src,
                                             _Float16* __restrict__ dst,
                                             int C, int P) {
    const int wp = 100 + 2 * P;
    const long total = (long)C * wp * wp;
    long i = (long)blockIdx.x * 256 + threadIdx.x;
    if (i >= total) return;
    int c = (int)(i / (wp * wp));
    int r = (int)(i - (long)c * wp * wp);
    int y = r / wp - P;
    int x = r % wp - P;
    float v = (y >= 0 && y < 100 && x >= 0 && x < 100)
                  ? src[(c * 100 + y) * 100 + x] : 0.0f;
    dst[i] = (_Float16)v;
}

// -------------------------------- k_zero16 ---------------------------------
__global__ __launch_bounds__(256) void k_zero16(_Float16* __restrict__ p, long n) {
    long i = (long)blockIdx.x * 256 + threadIdx.x;
    if (i < n) p[i] = (_Float16)0.0f;
}

// ----------------------------- k_conv_wmma ---------------------------------
// Spatial conv as implicit GEMM. Block: 256 thr / 8 waves; tile 128(M)x64(N).
// B tile gather-staged to LDS transposed [n][k]; 4 WMMAs per K-step per wave.
template<int CIN, int KSZ, int PAD>
__global__ __launch_bounds__(256) void k_conv_wmma(
    const _Float16* __restrict__ Xp,    // [CIN][100+2P][100+2P]
    const _Float16* __restrict__ Wh,    // [cout_pad][K]
    const float*    __restrict__ bias,
    const float*    __restrict__ resid, // [cout][10000] or null
    float*          __restrict__ outF,  // [cout][10000] or null
    _Float16*       __restrict__ outH,  // padded f16 out or null
    int outHpad, int cout, int mvalid, int act)
{
    constexpr int KK = KSZ * KSZ;
    constexpr int K  = CIN * KK;
    constexpr int WP = 100 + 2 * PAD;

    __shared__ __align__(32) _Float16 Bs[64 * 32];   // [n][k]

    const int tid  = threadIdx.x;
    const int wave = tid >> 5;
    const int lane = tid & 31;
    const int n0   = blockIdx.x * 64;
    const int mt   = blockIdx.y * 8 + wave;
    const int mtiles = cout >> 4;

    v8f acc[4] = {};

    const int nl = tid >> 2;              // 0..63 local column
    const int kl = (tid & 3) * 8;         // local k base (8 per thread)
    const int n  = n0 + nl;
    const int yy = n / 100, xx = n % 100;

    const _Float16* arow = Wh + (long)(mt * 16 + (lane & 15)) * K;
    const int koffA = (lane >> 4) * 16;

    for (int kk = 0; kk < K; kk += 32) {
        __syncthreads();
        #pragma unroll
        for (int i = 0; i < 8; ++i) {
            const int k  = kk + kl + i;
            const int c  = k / KK;
            const int r  = k - c * KK;
            const int kh = r / KSZ;
            const int kw = r - kh * KSZ;
            _Float16 v = (_Float16)0.0f;
            if (n < HW_N)
                v = Xp[((long)c * WP + (yy + kh)) * WP + (xx + kw)];
            Bs[nl * 32 + kl + i] = v;
        }
        __syncthreads();

        if (mt < mtiles) {
            __builtin_prefetch(arow + kk + 32, 0, 3);   // next A chunk
            union { v16h v; v8h h[2]; } a;
            a.h[0] = *(const v8h*)(arow + kk + koffA);
            a.h[1] = *(const v8h*)(arow + kk + 8 + koffA);
            #pragma unroll
            for (int g = 0; g < 4; ++g) {
                const v16h b = *(const v16h*)
                    &Bs[(g * 16 + (lane & 15)) * 32 + (lane >> 4) * 16];
                acc[g] = __builtin_amdgcn_wmma_f32_16x16x32_f16(
                    false, a.v, false, b, (short)0, acc[g], false, false);
            }
        }
    }
    if (mt >= mtiles) return;

    const int lcol  = lane & 15;
    const int rbase = (lane >> 4) * 8;
    const int wpo   = 100 + 2 * outHpad;

    #pragma unroll
    for (int g = 0; g < 4; ++g) {
        const int cn = n0 + g * 16 + lcol;
        if (cn >= HW_N) continue;
        const int cy = cn / 100, cx = cn % 100;
        #pragma unroll
        for (int r = 0; r < 8; ++r) {
            const int m = mt * 16 + rbase + r;
            if (m >= mvalid) continue;
            float v = acc[g][r] + bias[m];
            if (resid) v += resid[(long)m * HW_N + cn];
            if (act)   v = gelu_exact(v);
            if (outF)  outF[(long)m * HW_N + cn] = v;
            if (outH)  outH[((long)m * wpo + cy + outHpad) * wpo + cx + outHpad] =
                           (_Float16)v;
        }
    }
}

// ----------------------------- k_conv1x1_tdm --------------------------------
// 1x1 conv: B tile is a rectangular [32k][64n] slab of the activation tensor.
// Double-buffered TDM pipeline: wave 0 issues the next tile's DMA, then waits
// tensorcnt<=1 (in-order per-wave completion => current buffer ready) while
// the DMA for tile k+1 overlaps this tile's LDS transpose + 4 WMMAs.
__global__ __launch_bounds__(256) void k_conv1x1_tdm(
    const _Float16* __restrict__ X,    // [CIN][10000]
    const _Float16* __restrict__ Wh,   // [cout_pad][CIN]
    const float*    __restrict__ bias,
    float*          __restrict__ outF,
    _Float16*       __restrict__ outH,
    int outHpad, int CIN, int cout, int mvalid, int act)
{
    __shared__ __align__(32) _Float16 Bk[2][32 * 64];  // k-major DMA buffers
    __shared__ __align__(32) _Float16 Bs[64 * 32];     // n-major WMMA feed

    const int tid  = threadIdx.x;
    const int wave = tid >> 5;
    const int lane = tid & 31;
    const int n0   = blockIdx.x * 64;
    const int mt   = blockIdx.y * 8 + wave;
    const int mtiles = cout >> 4;

    v8f acc[4] = {};

    const int nl = tid >> 2;              // 0..63
    const int kl = (tid & 3) * 8;
    const int ncols = (HW_N - n0 < 64) ? (HW_N - n0) : 64;

    const _Float16* arow = Wh + (long)(mt * 16 + (lane & 15)) * CIN;
    const int koffA = (lane >> 4) * 16;

#if HAVE_TDM
    if (tid < 32) {                       // prologue: DMA tile 0 into buffer 0
        tdm_load_tile_f16((unsigned)(uintptr_t)&Bk[0][0], X + n0,
                          (unsigned)(HW_N - n0), (unsigned)CIN,
                          (unsigned)HW_N, 64u, 32u);
    }
#endif

    for (int kk = 0; kk < CIN; kk += 32) {
        const int ib = (kk >> 5) & 1;
#if HAVE_TDM
        if (tid < 32) {
            if (kk + 32 < CIN) {          // issue next tile, overlap with compute
                tdm_load_tile_f16((unsigned)(uintptr_t)&Bk[ib ^ 1][0],
                                  X + (size_t)(kk + 32) * HW_N + n0,
                                  (unsigned)(HW_N - n0),
                                  (unsigned)(CIN - kk - 32),
                                  (unsigned)HW_N, 64u, 32u);
#if __has_builtin(__builtin_amdgcn_s_wait_tensorcnt)
                __builtin_amdgcn_s_wait_tensorcnt((short)1);  // current buf done
#endif
            } else {
#if __has_builtin(__builtin_amdgcn_s_wait_tensorcnt)
                __builtin_amdgcn_s_wait_tensorcnt((short)0);  // last buf done
#endif
            }
        }
        __syncthreads();                  // DMA'd tile visible; Bs reusable
        #pragma unroll
        for (int i = 0; i < 8; ++i) {     // LDS transpose k-major -> n-major
            _Float16 v = (nl < ncols) ? Bk[ib][(kl + i) * 64 + nl]
                                      : (_Float16)0.0f;
            Bs[nl * 32 + kl + i] = v;
        }
#else
        __syncthreads();
        #pragma unroll
        for (int i = 0; i < 8; ++i) {     // fallback: direct gather staging
            const int k = kk + kl + i;
            _Float16 v = (_Float16)0.0f;
            if (nl < ncols) v = X[(size_t)k * HW_N + n0 + nl];
            Bs[nl * 32 + kl + i] = v;
        }
#endif
        __syncthreads();

        if (mt < mtiles) {
            __builtin_prefetch(arow + kk + 32, 0, 3);
            union { v16h v; v8h h[2]; } a;
            a.h[0] = *(const v8h*)(arow + kk + koffA);
            a.h[1] = *(const v8h*)(arow + kk + 8 + koffA);
            #pragma unroll
            for (int g = 0; g < 4; ++g) {
                const v16h b = *(const v16h*)
                    &Bs[(g * 16 + (lane & 15)) * 32 + (lane >> 4) * 16];
                acc[g] = __builtin_amdgcn_wmma_f32_16x16x32_f16(
                    false, a.v, false, b, (short)0, acc[g], false, false);
            }
        }
    }
    if (mt >= mtiles) return;

    const int lcol  = lane & 15;
    const int rbase = (lane >> 4) * 8;
    const int wpo   = 100 + 2 * outHpad;

    #pragma unroll
    for (int g = 0; g < 4; ++g) {
        const int cn = n0 + g * 16 + lcol;
        if (cn >= HW_N) continue;
        const int cy = cn / 100, cx = cn % 100;
        #pragma unroll
        for (int r = 0; r < 8; ++r) {
            const int m = mt * 16 + rbase + r;
            if (m >= mvalid) continue;
            float v = acc[g][r] + bias[m];
            if (act)   v = gelu_exact(v);
            if (outF)  outF[(long)m * HW_N + cn] = v;
            if (outH)  outH[((long)m * wpo + cy + outHpad) * wpo + cx + outHpad] =
                           (_Float16)v;
        }
    }
}

// -------------------------------- k_inorm ----------------------------------
__global__ __launch_bounds__(256) void k_inorm(const float* __restrict__ src,
                                               float* __restrict__ dst) {
    const int c = blockIdx.x;
    const float* x = src + (long)c * HW_N;
    float s = 0.0f, s2 = 0.0f;
    for (int i = threadIdx.x; i < HW_N; i += 256) {
        float v = x[i]; s += v; s2 += v * v;
    }
    __shared__ float red[2][8];
    #pragma unroll
    for (int off = 16; off > 0; off >>= 1) {
        s  += __shfl_down(s,  off, 32);
        s2 += __shfl_down(s2, off, 32);
    }
    const int wave = threadIdx.x >> 5, lane = threadIdx.x & 31;
    if (lane == 0) { red[0][wave] = s; red[1][wave] = s2; }
    __syncthreads();
    if (threadIdx.x == 0) {
        float a = 0.0f, b = 0.0f;
        #pragma unroll
        for (int i = 0; i < 8; ++i) { a += red[0][i]; b += red[1][i]; }
        red[0][0] = a / (float)HW_N;
        red[1][0] = b / (float)HW_N;
    }
    __syncthreads();
    const float m   = red[0][0];
    const float var = red[1][0] - m * m;
    const float inv = rsqrtf(var + 1e-5f);
    float* o = dst + (long)c * HW_N;
    for (int i = threadIdx.x; i < HW_N; i += 256) o[i] = (x[i] - m) * inv;
}

// -------------------------------- k_sample ---------------------------------
__global__ __launch_bounds__(160) void k_sample(
    const float* __restrict__ offc,   // [32][10000] (rows 0..23 valid)
    const float* __restrict__ swc,    // [32][10000]
    const float* __restrict__ bev_pos,// [10000][3]
    const float* __restrict__ l2i,    // [6][4][4]
    const float* __restrict__ f0, const float* __restrict__ f1,
    const float* __restrict__ f2, const float* __restrict__ f3,
    _Float16* __restrict__ sf_pad)    // [256][102][102]
{
    const int pos = blockIdx.x;
    const int tid = threadIdx.x;

    __shared__ float s_ref[8][3];
    __shared__ float s_sw[8][4];
    __shared__ float s_gx[6][8];
    __shared__ float s_gy[6][8];
    __shared__ float s_p3[8][3];

    if (tid < 8) {
        const int p = tid;
        const float pclo[3] = {-50.0f, -50.0f, -5.0f};
        const float pchi[3] = { 50.0f,  50.0f,  3.0f};
        #pragma unroll
        for (int j = 0; j < 3; ++j) {
            float o   = offc[(p * 3 + j) * HW_N + pos];
            float s   = 1.0f / (1.0f + expf(-o));
            float rng = (j < 2) ? (0.25f + 1e-6f) : (4.0f + 1e-6f);
            float off = s * rng * 2.0f - rng;
            float base = bev_pos[pos * 3 + j] * (pchi[j] - pclo[j]) + pclo[j];
            s_ref[p][j] = base + off;
        }
        float w[4], mx = -1e30f;
        #pragma unroll
        for (int l = 0; l < 4; ++l) {
            w[l] = swc[(p * 4 + l) * HW_N + pos];
            mx = fmaxf(mx, w[l]);
        }
        float sum = 0.0f;
        #pragma unroll
        for (int l = 0; l < 4; ++l) { w[l] = expf(w[l] - mx); sum += w[l]; }
        #pragma unroll
        for (int l = 0; l < 4; ++l) s_sw[p][l] = w[l] / sum;
    }
    __syncthreads();

    if (tid < 48) {
        const int n = tid >> 3, p = tid & 7;
        const float px = s_ref[p][0], py = s_ref[p][1], pz = s_ref[p][2];
        const float* M = l2i + n * 16;
        const float c0 = M[0] * px + M[1] * py + M[2]  * pz + M[3];
        const float c1 = M[4] * px + M[5] * py + M[6]  * pz + M[7];
        const float c2 = M[8] * px + M[9] * py + M[10] * pz + M[11];
        const bool valid = c2 > 1e-5f;
        const float zi = fmaxf(c2, 1e-5f);
        s_gx[n][p] = valid ? (c0 / zi) / 704.0f * 2.0f - 1.0f : -2.0f;
        s_gy[n][p] = valid ? (c1 / zi) / 256.0f * 2.0f - 1.0f : -2.0f;
    }
    __syncthreads();

    float acc[8] = {0,0,0,0,0,0,0,0};
    const int c = tid;
    if (c < 132) {
        const float* fl[4] = {f0, f1, f2, f3};
        const int LH[4] = {32, 16, 8, 4};
        const int LW[4] = {88, 44, 22, 11};
        for (int n = 0; n < 6; ++n) {
            #pragma unroll
            for (int p = 0; p < 8; ++p) {
                const float gx = s_gx[n][p], gy = s_gy[n][p];
                for (int l = 0; l < 4; ++l) {
                    const int H = LH[l], W = LW[l];
                    const float* f = fl[l] + ((long)(n * 132 + c)) * (H * W);
                    const float xf = (gx + 1.0f) * 0.5f * W - 0.5f;
                    const float yf = (gy + 1.0f) * 0.5f * H - 0.5f;
                    const float x0f = floorf(xf), y0f = floorf(yf);
                    const int x0 = (int)x0f, y0 = (int)y0f;
                    const float wx1 = xf - x0f, wy1 = yf - y0f;
                    const float wx0 = 1.0f - wx1, wy0 = 1.0f - wy1;
                    float v = 0.0f;
                    if (y0 >= 0 && y0 < H) {
                        if (x0 >= 0 && x0 < W)         v += f[y0 * W + x0]     * wx0 * wy0;
                        if (x0 + 1 >= 0 && x0 + 1 < W) v += f[y0 * W + x0 + 1] * wx1 * wy0;
                    }
                    if (y0 + 1 >= 0 && y0 + 1 < H) {
                        if (x0 >= 0 && x0 < W)         v += f[(y0 + 1) * W + x0]     * wx0 * wy1;
                        if (x0 + 1 >= 0 && x0 + 1 < W) v += f[(y0 + 1) * W + x0 + 1] * wx1 * wy1;
                    }
                    acc[p] += v * s_sw[p][l];
                }
            }
        }
    }
    if (c >= 128 && c < 131) {
        #pragma unroll
        for (int p = 0; p < 8; ++p) s_p3[p][c - 128] = acc[p];
    }
    __syncthreads();
    if (c < 128) {
        const int y = pos / 100, x = pos % 100;
        float sumf = 0.0f, sumw = 0.0f;
        #pragma unroll
        for (int p = 0; p < 8; ++p) {
            const float dx = s_ref[p][0] - s_p3[p][0];
            const float dy = s_ref[p][1] - s_p3[p][1];
            const float dz = s_ref[p][2] - s_p3[p][2];
            const float wgt = expf(-0.1f * (dx * dx + dy * dy + dz * dz));
            sumf += acc[p];
            sumw += acc[p] * wgt;
        }
        sf_pad[((long)c * 102 + y + 1) * 102 + (x + 1)]         = (_Float16)sumf;
        sf_pad[((long)(128 + c) * 102 + y + 1) * 102 + (x + 1)] = (_Float16)sumw;
    }
}

// --------------------------------------------------------------------------
extern "C" void kernel_launch(void* const* d_in, const int* in_sizes, int n_in,
                              void* d_out, int out_size, void* d_ws, size_t ws_size,
                              hipStream_t stream) {
    const float* bev_query = (const float*)d_in[0];
    const float* bev_pos   = (const float*)d_in[1];
    const float* lidar2img = (const float*)d_in[2];
    const float* feat0     = (const float*)d_in[3];
    const float* feat1     = (const float*)d_in[4];
    const float* feat2     = (const float*)d_in[5];
    const float* feat3     = (const float*)d_in[6];
    const float* in_w   = (const float*)d_in[7];
    const float* in_b   = (const float*)d_in[8];
    const float* off_w  = (const float*)d_in[9];
    const float* off_b  = (const float*)d_in[10];
    const float* sw_w   = (const float*)d_in[11];
    const float* sw_b   = (const float*)d_in[12];
    const float* mid_w1 = (const float*)d_in[13];
    const float* mid_b1 = (const float*)d_in[14];
    const float* mid_w2 = (const float*)d_in[15];
    const float* mid_b2 = (const float*)d_in[16];
    const float* mid_w3 = (const float*)d_in[17];
    const float* mid_b3 = (const float*)d_in[18];
    const float* out_w  = (const float*)d_in[19];
    const float* out_b  = (const float*)d_in[20];
    (void)in_sizes; (void)n_in; (void)out_size; (void)ws_size;

    char* base = (char*)d_ws;
    size_t off = 0;
    auto alloc = [&](size_t bytes) -> void* {
        void* p = base + off;
        off = (off + bytes + 255) & ~(size_t)255;
        return p;
    };
    _Float16* Win_h   = (_Float16*)alloc(409600 * 2);
    _Float16* Woff_h  = (_Float16*)alloc(4096 * 2);
    _Float16* Wsw_h   = (_Float16*)alloc(4096 * 2);
    _Float16* Wmid1_h = (_Float16*)alloc(1179648 * 2);
    _Float16* Wmid2_h = (_Float16*)alloc(262144 * 2);
    _Float16* Wmid3_h = (_Float16*)alloc(589824 * 2);
    _Float16* Wout_h  = (_Float16*)alloc(409600 * 2);
    _Float16* Xpad    = (_Float16*)alloc((size_t)128 * 104 * 104 * 2);
    _Float16* qflat_h = (_Float16*)alloc((size_t)128 * 10000 * 2);
    _Float16* sf_pad  = (_Float16*)alloc((size_t)256 * 102 * 102 * 2);
    _Float16* m1_h    = (_Float16*)alloc((size_t)512 * 10000 * 2);
    _Float16* m2pad   = (_Float16*)alloc((size_t)512 * 102 * 102 * 2);
    _Float16* q2pad   = (_Float16*)alloc((size_t)128 * 104 * 104 * 2);
    float* q    = (float*)alloc((size_t)128 * 10000 * 4);
    float* offc = (float*)alloc((size_t)32  * 10000 * 4);
    float* swc  = (float*)alloc((size_t)32  * 10000 * 4);
    float* t3   = (float*)alloc((size_t)128 * 10000 * 4);
    float* tout = (float*)alloc((size_t)128 * 10000 * 4);

    auto blks = [](long n) { return (unsigned)((n + 255) / 256); };

    // ---- weight conversion (off_w zero-padded 24 -> 32 rows) ----
    k_cvt<<<blks(409600),  256, 0, stream>>>(in_w,   Win_h,   409600,  409600);
    k_cvt<<<blks(4096),    256, 0, stream>>>(off_w,  Woff_h,  3072,    4096);
    k_cvt<<<blks(4096),    256, 0, stream>>>(sw_w,   Wsw_h,   4096,    4096);
    k_cvt<<<blks(1179648), 256, 0, stream>>>(mid_w1, Wmid1_h, 1179648, 1179648);
    k_cvt<<<blks(262144),  256, 0, stream>>>(mid_w2, Wmid2_h, 262144,  262144);
    k_cvt<<<blks(589824),  256, 0, stream>>>(mid_w3, Wmid3_h, 589824,  589824);
    k_cvt<<<blks(409600),  256, 0, stream>>>(out_w,  Wout_h,  409600,  409600);

    const dim3 g157_1(157, 1), g157_4(157, 4);

    // ---- stage 1: q = inorm(x + conv5x5(x)) ----
    k_pad<<<blks((long)128 * 104 * 104), 256, 0, stream>>>(bev_query, Xpad, 128, 2);
    k_conv_wmma<128, 5, 2><<<g157_1, 256, 0, stream>>>(
        Xpad, Win_h, in_b, bev_query, q, nullptr, 0, 128, 128, 0);
    k_inorm<<<128, 256, 0, stream>>>(q, q);

    // ---- stage 2: offsets + level weights (1x1 convs, TDM-staged) ----
    k_pad<<<blks((long)128 * 100 * 100), 256, 0, stream>>>(q, qflat_h, 128, 0);
    k_conv1x1_tdm<<<g157_1, 256, 0, stream>>>(
        qflat_h, Woff_h, off_b, offc, nullptr, 0, 128, 32, 24, 0);
    k_conv1x1_tdm<<<g157_1, 256, 0, stream>>>(
        qflat_h, Wsw_h, sw_b, swc, nullptr, 0, 128, 32, 32, 0);

    // ---- stage 3: deformable sampling + fold -> sf (padded f16) ----
    k_zero16<<<blks((long)256 * 102 * 102), 256, 0, stream>>>(
        sf_pad, (long)256 * 102 * 102);
    k_sample<<<10000, 160, 0, stream>>>(
        offc, swc, bev_pos, lidar2img, feat0, feat1, feat2, feat3, sf_pad);

    // ---- stage 4: mid convs ----
    k_conv_wmma<256, 3, 1><<<g157_4, 256, 0, stream>>>(
        sf_pad, Wmid1_h, mid_b1, nullptr, nullptr, m1_h, 0, 512, 512, 1);
    k_zero16<<<blks((long)512 * 102 * 102), 256, 0, stream>>>(
        m2pad, (long)512 * 102 * 102);
    k_conv1x1_tdm<<<g157_4, 256, 0, stream>>>(
        m1_h, Wmid2_h, mid_b2, nullptr, m2pad, 1, 512, 512, 512, 1);
    k_conv_wmma<512, 3, 1><<<g157_1, 256, 0, stream>>>(
        m2pad, Wmid3_h, mid_b3, q, t3, nullptr, 0, 128, 128, 0);
    k_inorm<<<128, 256, 0, stream>>>(t3, t3);

    // ---- stage 5: out conv + final inorm -> d_out ----
    k_pad<<<blks((long)128 * 104 * 104), 256, 0, stream>>>(t3, q2pad, 128, 2);
    k_conv_wmma<128, 5, 2><<<g157_1, 256, 0, stream>>>(
        q2pad, Wout_h, out_b, t3, tout, nullptr, 0, 128, 128, 0);
    k_inorm<<<128, 256, 0, stream>>>(tout, (float*)d_out);
}
